// LinearAttentionBlock_53764400611495
// MI455X (gfx1250) — compile-verified, matrix operand order checked by reading, not run on hardware
//
#include <hip/hip_runtime.h>
#include <hip/hip_bf16.h>

// ---------------------------------------------------------------------------
// LinearAttentionBlock on MI455X (gfx1250), wave32 + WMMA f16 + async-LDS.
//
//   K1: qkv = W_qkv @ x             (WMMA GEMM, f32 in -> f16 out)
//   K2: per-k-row max & sum(exp)    (f16 vector reads, f32 reduce)
//   K3: context[d,e] = sum_n exp(k-max)/Z * v    (small, VALU)
//   K4: W_eff[b] = W_out @ blockdiag(C^T)        (tiny, VALU)
//   K5: out = W_eff @ q + b_out     (WMMA GEMM, f16 B via ASYNC->LDS, f32 out)
//
// K5's B operand (q, f16) is staged with GLOBAL_LOAD_ASYNC_TO_LDS_B128 into a
// double-buffered raw LDS tile (tracked by ASYNCcnt), overlapping the WMMAs;
// falls back to register pipelining if the builtins are unavailable.
// ---------------------------------------------------------------------------

#if defined(__has_builtin)
#if __has_builtin(__builtin_amdgcn_global_load_async_to_lds_b128) && \
    __has_builtin(__builtin_amdgcn_s_wait_asynccnt)
#define ASYNC_LDS 1
#endif
#endif
#ifndef ASYNC_LDS
#define ASYNC_LDS 0
#endif

#if ASYNC_LDS
// builtin expects pointers to GCC-style int4 vectors in AS1 (global) / AS3 (LDS)
typedef int v4i_gcc __attribute__((vector_size(16)));
typedef __attribute__((address_space(1))) v4i_gcc* as1_v4p;
typedef __attribute__((address_space(3))) v4i_gcc* as3_v4p;
#endif

typedef _Float16 v4h  __attribute__((ext_vector_type(4)));
typedef _Float16 v8h  __attribute__((ext_vector_type(8)));
typedef _Float16 v16h __attribute__((ext_vector_type(16)));
typedef float    v8f  __attribute__((ext_vector_type(8)));

union FragAB { v16h v; v8h h[2]; };

#define LDS_STRIDE 40   // halves; 80B rows -> 16-distinct-bank lane pattern, 16B aligned

// D[m,n] = sum_k A[m,k]*B[k,n] (+ bias[m]).
// A: MxK row-major f32.  B: KxN row-major (f32 or f16).  D: f32 or f16.
// 128x128 tile, K chunked by 32, pipelined global->LDS staging.
template<bool BIAS, typename TB, typename TD, int NN, int KK>
__global__ __launch_bounds__(256)
void gemm_wmma_kernel(const float* __restrict__ Aall, long aStride,
                      const TB* __restrict__ Ball, long bStride,
                      TD* __restrict__ Dall, long dStride,
                      const float* __restrict__ bias)
{
    constexpr int  KCH    = KK / 32;
    constexpr bool BF16IN = (sizeof(TB) == 2);
    __shared__ __align__(16) _Float16 As[128 * LDS_STRIDE]; // [m][kk]
    __shared__ __align__(16) _Float16 Bs[128 * LDS_STRIDE]; // [n][kk] (transposed)
#if ASYNC_LDS
    __shared__ __align__(16) _Float16 Braw[2][32 * 128];    // raw [kk][n] f16 chunks
#endif

    const int batch = blockIdx.z;
    const float* A  = Aall + (long)batch * aStride;
    const TB*    Bg = Ball + (long)batch * bStride;
    TD*          D  = Dall + (long)batch * dStride;

    const int m0 = blockIdx.y * 128;
    const int n0 = blockIdx.x * 128;

    const int tid  = threadIdx.x;
    const int lane = tid & 31;
    const int wv   = tid >> 5;    // 0..7
    const int wm   = wv >> 1;     // 0..3 : 32-row slab
    const int wn   = wv & 1;      // 0..1 : 64-col slab
    const int g    = lane >> 4;   // half-wave group
    const int l16  = lane & 15;

    // ---- loop-invariant staging bases ----
    const int aRow = tid >> 3;
    const int aKK  = (tid & 7) << 2;
    const float* aBase = A + (long)(m0 + aRow) * KK + aKK;
    _Float16*    aLds  = &As[aRow * LDS_STRIDE + aKK];

    // B (f32 path): k-rows (tid>>5)+8i, cols (tid&31)*4
    // B (f16 path): k-rows (tid>>4)+16i, cols (tid&15)*8
    const int bKK32 = tid >> 5, bN4 = (tid & 31) << 2;
    const int bKK16 = tid >> 4, bN8 = (tid & 15) << 3;
    const TB* bBase = BF16IN
        ? (Bg + (long)bKK16 * NN + n0 + bN8)
        : (Bg + (long)bKK32 * NN + n0 + bN4);

    v8f acc[2][4];
    const v8f vzero = {0.f,0.f,0.f,0.f,0.f,0.f,0.f,0.f};
    #pragma unroll
    for (int i = 0; i < 2; ++i)
        #pragma unroll
        for (int j = 0; j < 4; ++j)
            acc[i][j] = vzero;

    // ---- prologue: stage chunk 0 ----
    float4 aReg[4];
    float4 bRegF[4];
    v8h    bRegH[2];
    (void)bRegH; (void)bRegF;
    #pragma unroll
    for (int i = 0; i < 4; ++i)
        aReg[i] = *reinterpret_cast<const float4*>(aBase + (long)i * 32 * KK);
    if constexpr (BF16IN) {
#if ASYNC_LDS
        #pragma unroll
        for (int i = 0; i < 2; ++i)
            __builtin_amdgcn_global_load_async_to_lds_b128(
                (as1_v4p)(bBase + (long)(16 * i) * NN),
                (as3_v4p)(&Braw[0][(bKK16 + 16 * i) * 128 + bN8]),
                0, 0);
#else
        #pragma unroll
        for (int i = 0; i < 2; ++i)
            bRegH[i] = *reinterpret_cast<const v8h*>(bBase + (long)(16 * i) * NN);
#endif
    } else {
        #pragma unroll
        for (int i = 0; i < 4; ++i)
            bRegF[i] = *reinterpret_cast<const float4*>(bBase + (long)(8 * i) * NN);
    }

    #pragma unroll 1
    for (int kc = 0; kc < KCH; ++kc) {
        // ---- A: registers -> LDS (f32 -> f16) ----
        #pragma unroll
        for (int i = 0; i < 4; ++i) {
            v4h hx = { (_Float16)aReg[i].x, (_Float16)aReg[i].y,
                       (_Float16)aReg[i].z, (_Float16)aReg[i].w };
            *reinterpret_cast<v4h*>(aLds + i * 32 * LDS_STRIDE) = hx;
        }
        // ---- B: materialize transposed chunk into Bs ----
        if constexpr (BF16IN) {
#if ASYNC_LDS
            // issue next chunk's async copies first (overlap with WMMAs below)
            if (kc + 1 < KCH) {
                const int k0n = (kc + 1) * 32;
                _Float16* dst = &Braw[(kc + 1) & 1][0];
                #pragma unroll
                for (int i = 0; i < 2; ++i)
                    __builtin_amdgcn_global_load_async_to_lds_b128(
                        (as1_v4p)(bBase + (long)(k0n + 16 * i) * NN),
                        (as3_v4p)(dst + (bKK16 + 16 * i) * 128 + bN8),
                        0, 0);
                __builtin_amdgcn_s_wait_asynccnt(2);  // current chunk landed
            } else {
                __builtin_amdgcn_s_wait_asynccnt(0);
            }
            const _Float16* src = &Braw[kc & 1][0];
            #pragma unroll
            for (int i = 0; i < 2; ++i) {
                const int kk = bKK16 + 16 * i;
                v8h hb = *reinterpret_cast<const v8h*>(src + kk * 128 + bN8);
                #pragma unroll
                for (int j = 0; j < 8; ++j)
                    Bs[(bN8 + j) * LDS_STRIDE + kk] = hb[j];
            }
#else
            #pragma unroll
            for (int i = 0; i < 2; ++i) {
                const int kk = bKK16 + 16 * i;
                #pragma unroll
                for (int j = 0; j < 8; ++j)
                    Bs[(bN8 + j) * LDS_STRIDE + kk] = bRegH[i][j];
            }
#endif
        } else {
            #pragma unroll
            for (int i = 0; i < 4; ++i) {
                const int kk = bKK32 + 8 * i;
                Bs[(bN4 + 0) * LDS_STRIDE + kk] = (_Float16)bRegF[i].x;
                Bs[(bN4 + 1) * LDS_STRIDE + kk] = (_Float16)bRegF[i].y;
                Bs[(bN4 + 2) * LDS_STRIDE + kk] = (_Float16)bRegF[i].z;
                Bs[(bN4 + 3) * LDS_STRIDE + kk] = (_Float16)bRegF[i].w;
            }
        }
        __syncthreads();

        // ---- prefetch next chunk's globals (overlap with WMMA below) ----
        if (kc + 1 < KCH) {
            const int k0n = (kc + 1) * 32;
            #pragma unroll
            for (int i = 0; i < 4; ++i)
                aReg[i] = *reinterpret_cast<const float4*>(
                    aBase + k0n + (long)i * 32 * KK);
            if constexpr (BF16IN) {
#if !ASYNC_LDS
                #pragma unroll
                for (int i = 0; i < 2; ++i)
                    bRegH[i] = *reinterpret_cast<const v8h*>(
                        bBase + (long)(k0n + 16 * i) * NN);
#endif
            } else {
                #pragma unroll
                for (int i = 0; i < 4; ++i)
                    bRegF[i] = *reinterpret_cast<const float4*>(
                        bBase + (long)(k0n + 8 * i) * NN);
            }
        }

        // ---- fragments: lane L -> row/col (L&15), K runs {8g..+7, 16+8g..+7} ----
        FragAB afr[2], bfr[4];
        #pragma unroll
        for (int i = 0; i < 2; ++i) {
            const int row = wm * 32 + i * 16 + l16;
            const _Float16* p = &As[row * LDS_STRIDE + 8 * g];
            afr[i].h[0] = *reinterpret_cast<const v8h*>(p);
            afr[i].h[1] = *reinterpret_cast<const v8h*>(p + 16);
        }
        #pragma unroll
        for (int j = 0; j < 4; ++j) {
            const int col = wn * 64 + j * 16 + l16;
            const _Float16* p = &Bs[col * LDS_STRIDE + 8 * g];
            bfr[j].h[0] = *reinterpret_cast<const v8h*>(p);
            bfr[j].h[1] = *reinterpret_cast<const v8h*>(p + 16);
        }
        #pragma unroll
        for (int i = 0; i < 2; ++i)
            #pragma unroll
            for (int j = 0; j < 4; ++j)
                acc[i][j] = __builtin_amdgcn_wmma_f32_16x16x32_f16(
                    false, afr[i].v, false, bfr[j].v,
                    (short)0, acc[i][j], false, false);
        __syncthreads();
    }

    // ---- epilogue: acc VGPR r = (M = r + 8g, N = l16) within 16x16 tile ----
    #pragma unroll
    for (int i = 0; i < 2; ++i) {
        const int rbase = m0 + wm * 32 + i * 16 + 8 * g;
        float bv[8];
        if (BIAS) {
            #pragma unroll
            for (int r = 0; r < 8; ++r) bv[r] = bias[rbase + r];
        }
        #pragma unroll
        for (int j = 0; j < 4; ++j) {
            const int col = n0 + wn * 64 + j * 16 + l16;
            TD* p = D + (long)rbase * NN + col;
            #pragma unroll
            for (int r = 0; r < 8; ++r) {
                float v = acc[i][j][r];
                if (BIAS) v += bv[r];
                p[(long)r * NN] = (TD)v;  // r*NN*size folds into IOFFSET
            }
        }
    }
}

// K2: per-row max and sum(exp) over n=16384 for the 2048 k rows (f16 input).
__global__ __launch_bounds__(256)
void softmax_stats_kernel(const _Float16* __restrict__ qkv,
                          float* __restrict__ kmax, float* __restrict__ ksum)
{
    __shared__ float red[256];
    const int row = blockIdx.x;           // 0..2047
    const int b  = row >> 8;
    const int oo = row & 255;
    const _Float16* kp = qkv + ((long)b * 768 + 256 + oo) * 16384;
    const int tid = threadIdx.x;

    float m = -1e30f;
    for (int i = tid * 8; i < 16384; i += 2048) {
        v8h h = *reinterpret_cast<const v8h*>(kp + i);
        #pragma unroll
        for (int j = 0; j < 8; ++j) m = fmaxf(m, (float)h[j]);
    }
    red[tid] = m; __syncthreads();
    for (int s = 128; s > 0; s >>= 1) {
        if (tid < s) red[tid] = fmaxf(red[tid], red[tid + s]);
        __syncthreads();
    }
    const float rowmax = red[0];
    __syncthreads();

    float sum = 0.f;
    for (int i = tid * 8; i < 16384; i += 2048) {
        v8h h = *reinterpret_cast<const v8h*>(kp + i);
        #pragma unroll
        for (int j = 0; j < 8; ++j) sum += __expf((float)h[j] - rowmax);
    }
    red[tid] = sum; __syncthreads();
    for (int s = 128; s > 0; s >>= 1) {
        if (tid < s) red[tid] += red[tid + s];
        __syncthreads();
    }
    if (tid == 0) { kmax[row] = rowmax; ksum[row] = red[0]; }
}

// K3: context[b,h,d,e] = (1/Z_d) sum_n exp(k[d,n]-max_d) * v[e,n]
__global__ __launch_bounds__(256)
void context_kernel(const _Float16* __restrict__ qkv,
                    const float* __restrict__ kmax, const float* __restrict__ ksum,
                    float* __restrict__ ctx)
{
    __shared__ float kt[32][132];
    __shared__ float vt[32][132];
    const int bh = blockIdx.x;           // 0..63
    const int b = bh >> 3, h = bh & 7;
    const _Float16* kp = qkv + ((long)b * 768 + 256 + h * 32) * 16384;
    const _Float16* vp = qkv + ((long)b * 768 + 512 + h * 32) * 16384;
    const float* km = kmax + b * 256 + h * 32;

    const int tid = threadIdx.x;
    const int d  = tid >> 3;             // 0..31
    const int e0 = (tid & 7) << 2;       // 0..28 step 4
    float acc[4] = {0.f, 0.f, 0.f, 0.f};

    for (int n0 = 0; n0 < 16384; n0 += 128) {
        __syncthreads();
        #pragma unroll
        for (int i = 0; i < 2; ++i) {
            int idx = tid + 256 * i;     // 0..511
            int dd = idx >> 4, nn8 = (idx & 15) << 3;
            v8h hk = *reinterpret_cast<const v8h*>(kp + (long)dd * 16384 + n0 + nn8);
            v8h hv = *reinterpret_cast<const v8h*>(vp + (long)dd * 16384 + n0 + nn8);
            const float kmv = km[dd];
            #pragma unroll
            for (int j = 0; j < 8; ++j) {
                kt[dd][nn8 + j] = __expf((float)hk[j] - kmv);
                vt[dd][nn8 + j] = (float)hv[j];
            }
        }
        __syncthreads();
        for (int nn = 0; nn < 128; ++nn) {
            float p = kt[d][nn];
            acc[0] += p * vt[e0 + 0][nn];
            acc[1] += p * vt[e0 + 1][nn];
            acc[2] += p * vt[e0 + 2][nn];
            acc[3] += p * vt[e0 + 3][nn];
        }
    }
    const float z = 1.f / ksum[b * 256 + h * 32 + d];
    float* cp = ctx + ((long)bh * 32 + d) * 32 + e0;
    cp[0] = acc[0] * z; cp[1] = acc[1] * z;
    cp[2] = acc[2] * z; cp[3] = acc[3] * z;
}

// K4: W_eff[b][o][h*32+d] = sum_e w_out[o][h*32+e] * ctx[b][h][d][e]
__global__ __launch_bounds__(256)
void weff_kernel(const float* __restrict__ w_out, const float* __restrict__ ctx,
                 float* __restrict__ weff)
{
    const long gid = (long)blockIdx.x * 256 + threadIdx.x;  // 0..524287
    const int b  = (int)(gid >> 16);
    const int o  = (int)(gid >> 8) & 255;
    const int hd = (int)gid & 255;
    const int h = hd >> 5, d = hd & 31;
    const float* wrow = w_out + (long)o * 256 + h * 32;
    const float* crow = ctx + (((long)b * 8 + h) * 32 + d) * 32;
    float s = 0.f;
    for (int e = 0; e < 32; ++e) s += wrow[e] * crow[e];
    weff[gid] = s;
}

extern "C" void kernel_launch(void* const* d_in, const int* in_sizes, int n_in,
                              void* d_out, int out_size, void* d_ws, size_t ws_size,
                              hipStream_t stream)
{
    const float* x     = (const float*)d_in[0];  // (8,256,128,128)
    const float* w_qkv = (const float*)d_in[1];  // (768,256)
    const float* w_out = (const float*)d_in[2];  // (256,256)
    const float* b_out = (const float*)d_in[3];  // (256,)
    float* out = (float*)d_out;                  // (8,256,128,128)

    // workspace layout: f16 qkv first, then f32 scratch
    _Float16* qkv = (_Float16*)d_ws;                    // 8*768*16384 halves
    float* fws  = (float*)(qkv + 100663296L);
    float* kmax = fws;                                  // 2048
    float* ksum = kmax + 2048;                          // 2048
    float* ctx  = ksum + 2048;                          // 65536
    float* weff = ctx + 65536;                          // 524288

    const long nb = 16384L;
    const long xStride   = 256L * nb;   // x / out batch strides
    const long qkvStride = 768L * nb;

    // K1: qkv(f16) = W_qkv @ x
    {
        dim3 grid(128, 6, 8);   // n-tiles, m-tiles(768/128), batch
        gemm_wmma_kernel<false, float, _Float16, 16384, 256><<<grid, 256, 0, stream>>>(
            w_qkv, 0L, x, xStride, qkv, qkvStride, nullptr);
    }
    // K2: softmax stats over k rows
    softmax_stats_kernel<<<2048, 256, 0, stream>>>(qkv, kmax, ksum);
    // K3: per-(b,h) context matrix
    context_kernel<<<64, 256, 0, stream>>>(qkv, kmax, ksum, ctx);
    // K4: effective output weights
    weff_kernel<<<2048, 256, 0, stream>>>(w_out, ctx, weff);
    // K5: out = W_eff @ q + b_out   (q = qkv rows 0..255 per batch, f16)
    {
        dim3 grid(128, 2, 8);   // n-tiles, m-tiles(256/128), batch
        gemm_wmma_kernel<true, _Float16, float, 16384, 256><<<grid, 256, 0, stream>>>(
            weff, 65536L, qkv, qkvStride, out, xStride, b_out);
    }
}
